// ModulatedGraphConv_16423954940708
// MI455X (gfx1250) — matrix operand order repaired and checked by reading, not compiled, not run on hardware
//
#include <hip/hip_runtime.h>

// ---------------------------------------------------------------------------
// ModulatedGraphConv for MI455X (gfx1250, wave32, WMMA).
//   out[b,i,d] = A[i,i]*M[i,d]*(x@W0)[b,i,d]
//              + sum_{j!=i} A[i,j]*M[j,d]*(x@W1)[b,j,d] + bias[d]
//   A = 0.5*((adj+adj2) + (adj+adj2)^T),  B=8192, N=17, C=256.
//
// GEMMs run as bf16x3 split-precision WMMA (hi*hi + hi*lo + lo*hi, f32 acc).
// The 17x17 node aggregation runs as a second WMMA (f32 16x16x4) with the
// diagonal (W0) path folded into an extended 32x36 coefficient matrix.
// ---------------------------------------------------------------------------

typedef __bf16 bf16_t;
typedef __attribute__((ext_vector_type(8)))  bf16_t v8bf;
typedef __attribute__((ext_vector_type(16))) bf16_t v16bf;
typedef __attribute__((ext_vector_type(8)))  float  v8f;
typedef __attribute__((ext_vector_type(4)))  float  v4f;
typedef __attribute__((ext_vector_type(2)))  float  v2f;

#define B_TOT   8192
#define NNODE   17
#define CIN     256
#define COUT    256
#define BT      16          // batches per block
#define KROWS   36          // 17 (M*h1 rows) + 17 (M*h0 rows) + 2 zero pad
#define BPITCH  2056        // 16*128 cols + 8 pad (f32) -> bank-conflict-free
#define XPITCH  264         // 256 + 8 pad (bf16)

__device__ __forceinline__ v16bf cat8(v8bf a, v8bf b) {
  return __builtin_shufflevector(a, b, 0,1,2,3,4,5,6,7,8,9,10,11,12,13,14,15);
}
__device__ __forceinline__ v8f wmma_bf16(v16bf a, v16bf b, v8f c) {
  return __builtin_amdgcn_wmma_f32_16x16x32_bf16(false, a, false, b, (short)0, c,
                                                 false, false);
}

// --- setup: Wt_{hi,lo}[w][d][c] = bf16 hi/lo split of W[w][c][d] (transposed
// so B-fragment rows are contiguous 16B chunks; stays hot in L2: 512 KB). ----
__global__ void mgc_setup_w(const float* __restrict__ W, bf16_t* __restrict__ wt) {
  int t = blockIdx.x * 256 + threadIdx.x;          // 0..131071
  int w = t >> 16, rem = t & 65535;
  int d = rem >> 8, c = rem & 255;
  float v  = W[w * 65536 + c * 256 + d];
  bf16_t h = (bf16_t)v;
  bf16_t l = (bf16_t)(v - (float)h);
  wt[(size_t)(2 * w + 0) * 65536 + d * 256 + c] = h;
  wt[(size_t)(2 * w + 1) * 65536 + d * 256 + c] = l;
}

// --- setup: Cext[32][36]: cols 0..16 -> off-diagonal A_sym (W1 path),
// cols 17..33 -> diag(A) on k-17==i (W0 path), rest zero padding. ------------
__global__ void mgc_setup_c(const float* __restrict__ adj,
                            const float* __restrict__ adj2,
                            float* __restrict__ cext) {
  for (int idx = threadIdx.x; idx < 32 * 36; idx += 256) {
    int i = idx / 36, k = idx % 36;
    float v = 0.0f;
    if (i < NNODE) {
      if (k < NNODE) {
        int j = k;
        if (j != i)
          v = 0.5f * ((adj[i * NNODE + j] + adj2[i * NNODE + j]) +
                      (adj[j * NNODE + i] + adj2[j * NNODE + i]));
      } else if (k < 2 * NNODE) {
        int j = k - NNODE;
        if (j == i) v = adj[i * NNODE + i] + adj2[i * NNODE + i];
      }
    }
    cext[idx] = v;
  }
}

// --- main fused kernel: block = 16 batches x 128 cols x all 17 nodes --------
__global__ __launch_bounds__(256) void mgc_main(
    const float* __restrict__ x, const bf16_t* __restrict__ wt,
    const float* __restrict__ cext, const float* __restrict__ M,
    const float* __restrict__ bias, float* __restrict__ out) {
  extern __shared__ char smem[];
  float*  Bbuf = (float*)smem;                               // [36][2056] f32
  bf16_t* xhi  = (bf16_t*)(smem + (size_t)KROWS * BPITCH * 4);  // [16][264]
  bf16_t* xlo  = xhi + 16 * XPITCH;

  const int tid  = threadIdx.x;
  const int lane = tid & 31;
  const int wv   = tid >> 5;
  const int hs   = lane >> 4;      // half-wave select
  const int l15  = lane & 15;
  const int cg   = blockIdx.x & 1;              // column group (2 x 128)
  const int b0   = (blockIdx.x >> 1) * BT;      // batch tile base

  // zero the two K-padding rows (34,35) of the aggregation B operand
  for (int idx = tid; idx < 2 * BPITCH; idx += 256)
    Bbuf[34 * BPITCH + idx] = 0.0f;

  const int matW      = (wv < 4) ? 1 : 0;       // waves 0-3: W1, 4-7: W0
  const int ct0       = (wv & 3) * 2;           // two 16-col tiles per wave
  const bf16_t* wHi   = wt + (size_t)(2 * matW + 0) * 65536;
  const bf16_t* wLo   = wt + (size_t)(2 * matW + 1) * 65536;

  for (int j = 0; j < NNODE; ++j) {
    __syncthreads();   // previous iteration's readers done before restage

    // ---- stage x[:, j, :] into LDS with hi/lo bf16 split ------------------
    {
      int row = tid >> 4;
      int c0  = (tid & 15) * 16;
      const float* src = x + ((size_t)(b0 + row) * NNODE + j) * CIN + c0;
      if (j + 1 < NNODE) __builtin_prefetch(src + CIN, 0, 1);
      float vals[16];
#pragma unroll
      for (int q = 0; q < 4; ++q) {
        v4f v = ((const v4f*)src)[q];
#pragma unroll
        for (int e = 0; e < 4; ++e) vals[4 * q + e] = v[e];
      }
      v8bf h0v, h1v, l0v, l1v;
#pragma unroll
      for (int e = 0; e < 8; ++e) {
        float f = vals[e];     bf16_t h = (bf16_t)f;
        h0v[e] = h; l0v[e] = (bf16_t)(f - (float)h);
        float g = vals[8 + e]; bf16_t k = (bf16_t)g;
        h1v[e] = k; l1v[e] = (bf16_t)(g - (float)k);
      }
      *(v8bf*)(xhi + row * XPITCH + c0)     = h0v;
      *(v8bf*)(xhi + row * XPITCH + c0 + 8) = h1v;
      *(v8bf*)(xlo + row * XPITCH + c0)     = l0v;
      *(v8bf*)(xlo + row * XPITCH + c0 + 8) = l1v;
    }
    __syncthreads();

    // ---- 16x16x256 bf16x3 GEMM: two column tiles per wave -----------------
    v8f acc0 = {}, acc1 = {};
    const int dg0 = cg * 128 + ct0 * 16 + l15;  // tile0 column (global d)
#pragma unroll
    for (int kc = 0; kc < 8; ++kc) {
      // A fragments (contiguous k order, same as B: lanes<16 hold K0-15, lanes>=16 K16-31)
      const int kb = kc * 32 + hs * 16;
      const bf16_t* ah = xhi + l15 * XPITCH + kb;
      const bf16_t* al = xlo + l15 * XPITCH + kb;
      v16bf ahi = cat8(*(const v8bf*)ah, *(const v8bf*)(ah + 8));
      v16bf alo = cat8(*(const v8bf*)al, *(const v8bf*)(al + 8));
      // B fragments (contiguous k order, same as A: lanes<16 hold K0-15, lanes>=16 K16-31)
      const int kbB = kc * 32 + hs * 16;
      const bf16_t* b0h = wHi + (size_t)dg0 * 256 + kbB;
      const bf16_t* b0l = wLo + (size_t)dg0 * 256 + kbB;
      const bf16_t* b1h = b0h + 16 * 256;
      const bf16_t* b1l = b0l + 16 * 256;
      v16bf B0h = cat8(*(const v8bf*)b0h, *(const v8bf*)(b0h + 8));
      v16bf B0l = cat8(*(const v8bf*)b0l, *(const v8bf*)(b0l + 8));
      v16bf B1h = cat8(*(const v8bf*)b1h, *(const v8bf*)(b1h + 8));
      v16bf B1l = cat8(*(const v8bf*)b1l, *(const v8bf*)(b1l + 8));
      acc0 = wmma_bf16(ahi, B0h, acc0);
      acc0 = wmma_bf16(ahi, B0l, acc0);
      acc0 = wmma_bf16(alo, B0h, acc0);
      acc1 = wmma_bf16(ahi, B1h, acc1);
      acc1 = wmma_bf16(ahi, B1l, acc1);
      acc1 = wmma_bf16(alo, B1h, acc1);
    }

    // ---- scale by M[j,d], deposit as aggregation-B row --------------------
    {
      const int rowB  = (matW == 1) ? j : (NNODE + j);
      const int dloc0 = ct0 * 16 + l15;
      const float m0 = M[j * COUT + cg * 128 + dloc0];
      const float m1 = M[j * COUT + cg * 128 + dloc0 + 16];
#pragma unroll
      for (int r = 0; r < 8; ++r) {
        int bp = r + hs * 8;  // batch-in-tile (C/D layout row)
        Bbuf[rowB * BPITCH + bp * 128 + dloc0]      = acc0[r] * m0;
        Bbuf[rowB * BPITCH + bp * 128 + dloc0 + 16] = acc1[r] * m1;
      }
    }
  }
  __syncthreads();

  // ---- phase 2: out[i, n] = Cext[32x36] x Bbuf[36x2048] via f32 WMMA ------
  v2f afr[2][9];
#pragma unroll
  for (int mt = 0; mt < 2; ++mt)
#pragma unroll
    for (int ks = 0; ks < 9; ++ks) {
      int i  = mt * 16 + l15;
      int k0 = ks * 4 + hs * 2;          // f32 A 16x4: v0=K k0, v1=K k0+1
      afr[mt][ks][0] = cext[i * 36 + k0];
      afr[mt][ks][1] = cext[i * 36 + k0 + 1];
    }

  for (int nt = wv * 16; nt < wv * 16 + 16; ++nt) {
    v8f cacc0 = {}, cacc1 = {};
#pragma unroll
    for (int ks = 0; ks < 9; ++ks) {
      int k0 = ks * 4 + hs * 2;
      v2f bfr;
      bfr[0] = Bbuf[k0 * BPITCH + nt * 16 + l15];
      bfr[1] = Bbuf[(k0 + 1) * BPITCH + nt * 16 + l15];
      cacc0 = __builtin_amdgcn_wmma_f32_16x16x4_f32(false, afr[0][ks], false, bfr,
                                                    (short)0, cacc0, false, false);
      cacc1 = __builtin_amdgcn_wmma_f32_16x16x4_f32(false, afr[1][ks], false, bfr,
                                                    (short)0, cacc1, false, false);
    }
    // ---- epilogue: + bias, coalesced stores (cols contiguous in d) --------
    const int n    = nt * 16 + l15;
    const int bp   = n >> 7;
    const int dloc = n & 127;
    const float bv = bias[cg * 128 + dloc];
    const size_t obase =
        (size_t)(b0 + bp) * NNODE * COUT + cg * 128 + dloc;
#pragma unroll
    for (int r = 0; r < 8; ++r) {
      int i0 = r + hs * 8;                       // rows 0..15: always valid
      out[obase + (size_t)i0 * COUT] = cacc0[r] + bv;
      int i1 = 16 + i0;                          // only i==16 is a real node
      if (i1 == 16) out[obase + (size_t)i1 * COUT] = cacc1[r] + bv;
    }
  }
}

extern "C" void kernel_launch(void* const* d_in, const int* in_sizes, int n_in,
                              void* d_out, int out_size, void* d_ws, size_t ws_size,
                              hipStream_t stream) {
  const float* x    = (const float*)d_in[0];
  const float* W    = (const float*)d_in[1];
  const float* M    = (const float*)d_in[2];
  const float* adj  = (const float*)d_in[3];
  const float* adj2 = (const float*)d_in[4];
  const float* bias = (const float*)d_in[5];
  float* out = (float*)d_out;

  bf16_t* wt  = (bf16_t*)d_ws;                                   // 4 x 128 KB
  float* cext = (float*)((char*)d_ws + 4 * (size_t)65536 * sizeof(bf16_t));

  mgc_setup_w<<<512, 256, 0, stream>>>(W, wt);
  mgc_setup_c<<<1, 256, 0, stream>>>(adj, adj2, cext);

  const size_t shmem = (size_t)KROWS * BPITCH * sizeof(float) +
                       2 * 16 * XPITCH * sizeof(bf16_t);         // 312,960 B
  mgc_main<<<(B_TOT / BT) * 2, 256, shmem, stream>>>(x, wt, cext, M, bias, out);
}
